// LiGRUModel_79018808312535
// MI455X (gfx1250) — matrix operand order repaired and checked by reading, not compiled, not run on hardware
//
#include <hip/hip_runtime.h>

// ---------------------------------------------------------------- types
typedef unsigned int   u32;
typedef unsigned short u16;
typedef __attribute__((ext_vector_type(16))) __bf16 v16bf;
typedef __attribute__((ext_vector_type(8)))  float  v8f;
typedef __attribute__((ext_vector_type(4)))  unsigned int v4u;
typedef __attribute__((ext_vector_type(8)))  int  v8i;
typedef __attribute__((ext_vector_type(4)))  int  v4i;

#define T_   500
#define B_   64
#define F_   89
#define H_   512
#define O_   72
#define B2   128          // 2*B
#define G_   1024         // 2*H
#define FP   96           // F padded to mult of 32
#define OP   80           // O padded to mult of 16
#define MTOT (T_*B2)      // 64000 rows in the big GEMMs
#define STR  20           // LDS row stride (dwords): 80B rows -> B128-aligned, low-conflict

// float -> bf16, round-to-nearest-even
__device__ __forceinline__ u16 f2bf(float f) {
    union { float f; u32 u; } v; v.f = f;
    u32 r = v.u + 0x7FFFu + ((v.u >> 16) & 1u);
    return (u16)(r >> 16);
}

// ------------------------------------------------- CDNA5 async memory->LDS (ASYNCcnt)
__device__ __forceinline__ void async_b128(u32* lds, const u32* g) {
    u32 l = (u32)(unsigned long long)lds;   // low 32 bits of generic ptr == LDS offset
    asm volatile("global_load_async_to_lds_b128 %0, %1, off" :: "v"(l), "v"(g) : "memory");
}
__device__ __forceinline__ void async_b64(u32* lds, const u32* g) {
    u32 l = (u32)(unsigned long long)lds;
    asm volatile("global_load_async_to_lds_b64 %0, %1, off" :: "v"(l), "v"(g) : "memory");
}
#define WAIT_ASYNC(imm) asm volatile("s_wait_asynccnt " imm ::: "memory")

// ------------------------------------------------- CDNA5 Tensor Data Mover (TENSORcnt)
// 2D tile load, data_size=8B. LDS padding: +16B after every 64B stored
// (pad_interval code 3 = 64B, pad_amount code 3 = 4 dwords) -> matches STR=20 rows
// when tile rows are 64B wide.
__device__ __forceinline__ void tdm_load_2d(u32 lds_off, const void* gp,
                                            u32 rows, u32 row_units, u32 stride_units) {
    unsigned long long ga = (unsigned long long)gp;
    v4u g0 = { 1u,                                   // count=1, user, no gather
               lds_off,                              // lds_addr
               (u32)ga,                              // global_addr[31:0]
               (u32)((ga >> 32) & 0x01FFFFFFu) | (2u << 30) };  // addr[56:32] | type=2
    v8i g1;
    g1[0] = (int)((3u << 16) | (1u << 20) | (3u << 22) | (3u << 25)); // 8B, pad on, 64B/16B
    g1[1] = (int)((stride_units & 0xFFFFu) << 16);                    // tensor_dim0 lo16
    g1[2] = (int)((stride_units >> 16) | ((rows & 0xFFFFu) << 16));   // td0 hi16 | td1 lo16
    g1[3] = (int)((row_units & 0xFFFFu) << 16);                       // td1 hi16(=0)|tile_dim0
    g1[4] = (int)(rows & 0xFFFFu);                                    // tile_dim1, tile_dim2=0
    g1[5] = (int)stride_units;                                        // tensor_dim0_stride lo32
    g1[6] = 0;                                                        // stride hi16 | td1_stride lo
    g1[7] = 0;
    v4i z4 = { 0, 0, 0, 0 };
    v8i z8 = { 0, 0, 0, 0, 0, 0, 0, 0 };
    __builtin_amdgcn_tensor_load_to_lds(g0, g1, z4, z4, z8, 0);  // 6-arg toolchain form
}

// ---------------------------------------------------------------- fragment loaders
// A-matrix 16x32 bf16 (contiguous k order used here, not the ISA table): lane m = lane&15, half = lane>>4
//   VGPR v : kpair = 8*half + v
__device__ __forceinline__ v16bf frag_a_lds(const u32* lds, int row, int half) {
    union { v16bf v; u32 u[8]; } f;
    const u32* p = lds + row * STR + half * 8;
#pragma unroll
    for (int i = 0; i < 4; ++i) f.u[i]     = p[i];
#pragma unroll
    for (int i = 0; i < 4; ++i) f.u[4 + i] = p[4 + i];
    return f.v;
}
// B-matrix 32x16 bf16: lane n = lane&15; lanes 0-15 hold K=0..15, 16-31 K=16..31
//   VGPR v : kpair = 8*half + v
__device__ __forceinline__ v16bf frag_b_lds(const u32* lds, int row, int half) {
    union { v16bf v; u32 u[8]; } f;
    const u32* p = lds + row * STR + half * 8;
#pragma unroll
    for (int i = 0; i < 8; ++i) f.u[i] = p[i];
    return f.v;
}
__device__ __forceinline__ v16bf frag_a_g(const u32* g, long row, int Ku, int k0u, int half) {
    union { v16bf v; u32 u[8]; } f;
    const u32* p = g + row * Ku + k0u + half * 8;
#pragma unroll
    for (int i = 0; i < 4; ++i) f.u[i]     = p[i];
#pragma unroll
    for (int i = 0; i < 4; ++i) f.u[4 + i] = p[4 + i];
    return f.v;
}
__device__ __forceinline__ v16bf frag_b_g(const u32* g, long row, int Ku, int k0u, int half) {
    union { v16bf v; u32 u[8]; } f;
    const u32* p = g + row * Ku + k0u + half * 8;
#pragma unroll
    for (int i = 0; i < 8; ++i) f.u[i] = p[i];
    return f.v;
}

// ---------------------------------------------------------------- weight convert (+pad)
__global__ void cvt_pad(const float* __restrict__ src, u16* __restrict__ dst,
                        int N, int K, int Np, int Kp) {
    int i = blockIdx.x * blockDim.x + threadIdx.x;
    if (i >= Np * Kp) return;
    int n = i / Kp, k = i % Kp;
    dst[i] = (n < N && k < K) ? f2bf(src[(long)n * K + k]) : (u16)0;
}

// ---------------------------------------------------------------- layer-0 input build
__global__ void build_x0(const float* __restrict__ batch, u16* __restrict__ xb) {
    long i = (long)blockIdx.x * blockDim.x + threadIdx.x;
    if (i >= (long)MTOT * FP) return;
    int k  = (int)(i % FP);
    long r = i / FP;
    int bp = (int)(r % B2);
    int t  = (int)(r / B2);
    int b  = (bp < B_) ? bp : bp - B_;
    int tt = (bp < B_) ? t  : (T_ - 1 - t);
    float v = (k < F_) ? batch[((long)b * T_ + tt) * F_ + k] : 0.f;
    xb[i] = f2bf(v);
}

// ---------------------------------------------------------------- next-layer input build
// y[b,t] = concat(hs[t][b], hs[T-1-t][B+b]) ; xb_next stacks y and time-reversed y
__global__ void build_next(const u16* __restrict__ hseq, u16* __restrict__ xb) {
    long i = (long)blockIdx.x * blockDim.x + threadIdx.x;
    if (i >= (long)MTOT * G_) return;
    int f  = (int)(i % G_);
    long r = i / G_;
    int bp = (int)(r % B2);
    int t  = (int)(r / B2);
    long src;
    if (bp < B_) {
        src = (f < H_) ? ((long)t * B2 + bp) * H_ + f
                       : ((long)(T_ - 1 - t) * B2 + (B_ + bp)) * H_ + (f - H_);
    } else {
        int b = bp - B_;
        src = (f < H_) ? ((long)(T_ - 1 - t) * B2 + b) * H_ + f
                       : ((long)t * B2 + bp) * H_ + (f - H_);
    }
    xb[i] = hseq[src];
}

// ---------------------------------------------------------------- big TN GEMM (gates)
// C[M,1024] = A[M,K](bf16) * W[1024,K](bf16)^T + bias
// WG tile 256x64, 8 waves; wave = 2 m-tiles x 4 n-tiles = 8 WMMA/chunk.
// Staging via Tensor Data Mover: wave 0 issues 2 descriptor loads per chunk,
// double-buffered, completion via s_wait_tensorcnt.
__global__ __launch_bounds__(256) void gemm_gates(
    const u16* __restrict__ A, const u16* __restrict__ W,
    const float* __restrict__ bias, float* __restrict__ C, int K) {
    __shared__ u32 As[2][256 * STR];
    __shared__ u32 Bs[2][64 * STR];
    const int Ku  = K >> 1;
    const int m0  = blockIdx.y * 256;
    const int n0  = blockIdx.x * 64;
    const int tid = threadIdx.x;
    const int wv  = tid >> 5, lane = tid & 31;
    const int half = lane >> 4, r16 = lane & 15;
    const u32* Ag = (const u32*)A;
    const u32* Wg = (const u32*)W;
    const u32* Abase = Ag + (long)m0 * Ku;   // tile origin (row m0, k=0)
    const u32* Bbase = Wg + (long)n0 * Ku;
    const u32  su = (u32)(Ku >> 1);          // row stride in 8B units

    auto stage = [&](int buf, int k0u) {
        if (wv == 0) {
            tdm_load_2d((u32)(unsigned long long)&As[buf][0], Abase + k0u, 256, 8, su);
            tdm_load_2d((u32)(unsigned long long)&Bs[buf][0], Bbase + k0u, 64,  8, su);
        }
    };

    v8f acc0[4], acc1[4];
#pragma unroll
    for (int i = 0; i < 4; ++i) {
        acc0[i] = (v8f){0.f,0.f,0.f,0.f,0.f,0.f,0.f,0.f};
        acc1[i] = (v8f){0.f,0.f,0.f,0.f,0.f,0.f,0.f,0.f};
    }

    const int NCH = Ku >> 4;
    stage(0, 0);
    for (int c = 0; c < NCH; ++c) {
        const int cur = c & 1;
        if (c + 1 < NCH) {
            stage(cur ^ 1, (c + 1) << 4);
            if (wv == 0) __builtin_amdgcn_s_wait_tensorcnt(2);
        } else {
            if (wv == 0) __builtin_amdgcn_s_wait_tensorcnt(0);
        }
        __syncthreads();
        v16bf a0 = frag_a_lds(As[cur], wv * 16 + r16, half);
        v16bf a1 = frag_a_lds(As[cur], 128 + wv * 16 + r16, half);
#pragma unroll
        for (int nt = 0; nt < 4; ++nt) {
            v16bf bf = frag_b_lds(Bs[cur], nt * 16 + r16, half);
            acc0[nt] = __builtin_amdgcn_wmma_f32_16x16x32_bf16(
                false, a0, false, bf, (short)0, acc0[nt], false, false);
            acc1[nt] = __builtin_amdgcn_wmma_f32_16x16x32_bf16(
                false, a1, false, bf, (short)0, acc1[nt], false, false);
        }
        __syncthreads();
    }
#pragma unroll
    for (int nt = 0; nt < 4; ++nt) {
        int n = n0 + nt * 16 + r16;
        float bn = bias[n];
#pragma unroll
        for (int v = 0; v < 8; ++v) {
            int m = m0 + wv * 16 + v + half * 8;
            C[(long)m * G_ + n]         = acc0[nt][v] + bn;
            C[(long)(m + 128) * G_ + n] = acc1[nt][v] + bn;
        }
    }
}

// ---------------------------------------------------------------- one recurrent step
// g = gates[t] + h_prev @ U^T ; h = sigm(g_z)*h + (1-sigm(g_z))*tanh(g_a)
// grid.x = 32 j-slices of 16; WG computes gate cols j and j+512 with fused epilogue.
// Staging via async-to-LDS (ASYNCcnt), double-buffered.
__global__ __launch_bounds__(256) void ligru_step(
    const u16* __restrict__ hprev,   // [128,512] bf16 (state t-1)
    const u16* __restrict__ U,       // [1024,512] bf16
    const float* __restrict__ gates, // [T*128,1024] fp32
    float* __restrict__ hf32,        // [128,512] fp32 state (in/out)
    u16* __restrict__ hnext,         // [128,512] bf16 (state t)
    u16* __restrict__ hseq_t,        // [128,512] slice of hseq at t
    int t) {
    __shared__ u32 As[2][128 * STR];
    __shared__ u32 Bs[2][32 * STR];
    const int j0  = blockIdx.x * 16;
    const int tid = threadIdx.x;
    const int wv  = tid >> 5, lane = tid & 31;
    const int half = lane >> 4, r16 = lane & 15;
    const u32* Ag = (const u32*)hprev;   // Ku = 256
    const u32* Ug = (const u32*)U;

    const int  ar = tid >> 1, ap = tid & 1;                 // A: 128 rows x 2 halves
    const u32* asrc = Ag + (long)ar * 256 + ap * 8;
    const int  br = tid >> 3, bp8 = tid & 7;                // B: 32 rows x 8 pairs (b64)
    const int  grow = (br < 16) ? (j0 + br) : (H_ + j0 + br - 16);
    const u32* bsrc = Ug + (long)grow * 256 + bp8 * 2;

    auto stage = [&](int buf, int k0) {
        u32* d = &As[buf][ar * STR + ap * 8];
        async_b128(d + 0, asrc + k0);
        async_b128(d + 4, asrc + k0 + 4);
        async_b64(&Bs[buf][br * STR + bp8 * 2], bsrc + k0);
    };

    v8f acc_a = (v8f){0.f,0.f,0.f,0.f,0.f,0.f,0.f,0.f};
    v8f acc_z = (v8f){0.f,0.f,0.f,0.f,0.f,0.f,0.f,0.f};

    stage(0, 0);
#pragma unroll 1
    for (int c = 0; c < 16; ++c) {
        const int cur = c & 1;
        if (c < 15) { stage(cur ^ 1, (c + 1) << 4); WAIT_ASYNC("0x3"); }
        else        { WAIT_ASYNC("0x0"); }
        __syncthreads();
        v16bf af  = frag_a_lds(As[cur], wv * 16 + r16, half);
        v16bf bfa = frag_b_lds(Bs[cur], r16, half);
        v16bf bfz = frag_b_lds(Bs[cur], 16 + r16, half);
        acc_a = __builtin_amdgcn_wmma_f32_16x16x32_bf16(
            false, af, false, bfa, (short)0, acc_a, false, false);
        acc_z = __builtin_amdgcn_wmma_f32_16x16x32_bf16(
            false, af, false, bfz, (short)0, acc_z, false, false);
        __syncthreads();
    }
    const float* g = gates + (long)t * B2 * G_;
    const int j = j0 + r16;
#pragma unroll
    for (int v = 0; v < 8; ++v) {
        int m = wv * 16 + v + half * 8;
        float ga = acc_a[v] + g[(long)m * G_ + j];
        float gz = acc_z[v] + g[(long)m * G_ + H_ + j];
        float z  = 1.f / (1.f + __expf(-gz));
        float hn = z * hf32[(long)m * H_ + j] + (1.f - z) * tanhf(ga);
        hf32[(long)m * H_ + j] = hn;
        u16 hb = f2bf(hn);
        hnext[(long)m * H_ + j]  = hb;
        hseq_t[(long)m * H_ + j] = hb;
    }
}

// ---------------------------------------------------------------- FC GEMM (wave/tile)
// C[M,80] = Y[M,1024](bf16) * Wfc[80,1024](bf16)^T  (bias applied in pack)
__global__ __launch_bounds__(32) void gemm_fc(
    const u16* __restrict__ Y, const u16* __restrict__ Wf, float* __restrict__ C) {
    const int m0 = blockIdx.y * 16, n0 = blockIdx.x * 16;
    const int lane = threadIdx.x & 31;
    const int half = lane >> 4, r16 = lane & 15;
    const u32* Yg = (const u32*)Y;   // Ku = 512
    const u32* Wg = (const u32*)Wf;
    v8f acc = (v8f){0.f,0.f,0.f,0.f,0.f,0.f,0.f,0.f};
    for (int k0 = 0; k0 < 512; k0 += 16) {
        v16bf af = frag_a_g(Yg, (long)(m0 + r16), 512, k0, half);
        v16bf bf = frag_b_g(Wg, (long)(n0 + r16), 512, k0, half);
        acc = __builtin_amdgcn_wmma_f32_16x16x32_bf16(
            false, af, false, bf, (short)0, acc, false, false);
    }
    int n = n0 + r16;
#pragma unroll
    for (int v = 0; v < 8; ++v) {
        int m = m0 + v + half * 8;
        C[(long)m * OP + n] = acc[v];
    }
}

// ---------------------------------------------------------------- pack valid frames
__global__ void pack_out(const float* __restrict__ outfc, const float* __restrict__ bfc,
                         const int* __restrict__ len, float* __restrict__ out) {
    int t = blockIdx.x, b = blockIdx.y, o = threadIdx.x;
    if (o >= O_) return;
    if (t >= len[b]) return;
    int p = 0;
    for (int i = 0; i < b; ++i) p += len[i];
    out[(long)(p + t) * O_ + o] = outfc[((long)t * B2 + b) * OP + o] + bfc[o];
}

// ---------------------------------------------------------------- host driver
extern "C" void kernel_launch(void* const* d_in, const int* in_sizes, int n_in,
                              void* d_out, int out_size, void* d_ws, size_t ws_size,
                              hipStream_t stream) {
    const float* batch   = (const float*)d_in[0];
    const int*   lengths = (const int*)  d_in[1];
    const float* w[3]  = {(const float*)d_in[2], (const float*)d_in[5], (const float*)d_in[8]};
    const float* bw[3] = {(const float*)d_in[3], (const float*)d_in[6], (const float*)d_in[9]};
    const float* u[3]  = {(const float*)d_in[4], (const float*)d_in[7], (const float*)d_in[10]};
    const float* wfc = (const float*)d_in[11];
    const float* bfc = (const float*)d_in[12];
    float* out = (float*)d_out;

    char* p = (char*)d_ws;
    auto alloc = [&](size_t bytes) -> void* {
        void* r = (void*)p;
        p += (bytes + 255) & ~(size_t)255;
        return r;
    };
    float* gates = (float*)alloc((size_t)MTOT * G_ * 4);   // 262 MB
    u16*   xb    = (u16*)  alloc((size_t)MTOT * G_ * 2);   // 131 MB (reused per layer)
    u16*   hseq  = (u16*)  alloc((size_t)MTOT * H_ * 2);   // 65.5 MB
    u16*   hb0   = (u16*)  alloc((size_t)B2 * H_ * 2);
    u16*   hb1   = (u16*)  alloc((size_t)B2 * H_ * 2);
    float* hf32  = (float*)alloc((size_t)B2 * H_ * 4);
    float* outfc = (float*)alloc((size_t)MTOT * OP * 4);   // 20.5 MB
    u16*   w0p   = (u16*)  alloc((size_t)G_ * FP * 2);
    u16*   w1b   = (u16*)  alloc((size_t)G_ * G_ * 2);
    u16*   w2b   = (u16*)  alloc((size_t)G_ * G_ * 2);
    u16*   u0b   = (u16*)  alloc((size_t)G_ * H_ * 2);
    u16*   u1b   = (u16*)  alloc((size_t)G_ * H_ * 2);
    u16*   u2b   = (u16*)  alloc((size_t)G_ * H_ * 2);
    u16*   wfcp  = (u16*)  alloc((size_t)OP * G_ * 2);

    auto cvt = [&](const float* s, u16* d, int N, int K, int Np, int Kp) {
        int n = Np * Kp;
        cvt_pad<<<(n + 255) / 256, 256, 0, stream>>>(s, d, N, K, Np, Kp);
    };
    cvt(w[0], w0p, G_, F_, G_, FP);
    cvt(w[1], w1b, G_, G_, G_, G_);
    cvt(w[2], w2b, G_, G_, G_, G_);
    cvt(u[0], u0b, G_, H_, G_, H_);
    cvt(u[1], u1b, G_, H_, G_, H_);
    cvt(u[2], u2b, G_, H_, G_, H_);
    cvt(wfc,  wfcp, O_, G_, OP, G_);

    {   // layer-0 input
        long n = (long)MTOT * FP;
        build_x0<<<(unsigned)((n + 255) / 256), 256, 0, stream>>>(batch, xb);
    }

    const u16* Wb[3] = {w0p, w1b, w2b};
    const u16* Ub[3] = {u0b, u1b, u2b};
    const int  Kin[3] = {FP, G_, G_};

    for (int l = 0; l < 3; ++l) {
        gemm_gates<<<dim3(G_ / 64, MTOT / 256), 256, 0, stream>>>(
            xb, Wb[l], bw[l], gates, Kin[l]);
        (void)hipMemsetAsync(hf32, 0, (size_t)B2 * H_ * 4, stream);
        (void)hipMemsetAsync(hb0,  0, (size_t)B2 * H_ * 2, stream);
        for (int t = 0; t < T_; ++t) {
            const u16* hp = (t & 1) ? hb1 : hb0;
            u16*       hn = (t & 1) ? hb0 : hb1;
            ligru_step<<<H_ / 16, 256, 0, stream>>>(
                hp, Ub[l], gates, hf32, hn, hseq + (size_t)t * B2 * H_, t);
        }
        {   // produce next layer input (or FC input after last layer)
            long n = (long)MTOT * G_;
            build_next<<<(unsigned)((n + 255) / 256), 256, 0, stream>>>(hseq, xb);
        }
    }

    gemm_fc<<<dim3(OP / 16, MTOT / 16), 32, 0, stream>>>(xb, wfcp, outfc);
    pack_out<<<dim3(T_, B_), 96, 0, stream>>>(outfc, bfc, lengths, out);
}
